// FeaturePropagationNeuralOperator_2989297238648
// MI455X (gfx1250) — compile-verified, matrix operand order checked by reading, not run on hardware
//
#include <hip/hip_runtime.h>
#include <hip/hip_bf16.h>

// ---------------------------------------------------------------------------
// FeaturePropagationNeuralOperator for MI455X (gfx1250, wave32, WMMA)
// B=4, N_SRC=2048, N_DST=8192, C_IN=256, C_SKIP=128, H=256, C_OUT=128, P=64, K=3
// ---------------------------------------------------------------------------

#define B_       4
#define N_SRC_   2048
#define N_DST_   8192
#define C_IN_    256
#define C_SKIP_  128
#define H_       256
#define C_OUT_   128
#define P_       64
#define M_TOTAL  (B_ * N_DST_)      // 32768 rows
#define K1_      (C_IN_ + C_SKIP_)  // 384
#define LDH_     264                // padded LDS row stride in halves (528B, 16B-aligned)

typedef _Float16 half_t;
typedef __attribute__((ext_vector_type(16))) _Float16 v16h;
typedef __attribute__((ext_vector_type(8)))  _Float16 v8h;
typedef __attribute__((ext_vector_type(8)))  float    v8f;

// Load a 16-half WMMA operand fragment (A-layout == B-layout, wave32):
// lane 0-15  : row/col = lane,     K = [k0..k0+7] , [k0+16..k0+23]
// lane 16-31 : row/col = lane-16,  K = [k0+8..+15], [k0+24..+31]
// Caller passes p already offset by (lane>>4)*8 halves; we grab 8 + 8 halves.
__device__ inline v16h load_frag16(const half_t* p) {
    v8h lo = *(const v8h*)p;
    v8h hi = *(const v8h*)(p + 16);
    return __builtin_shufflevector(lo, hi, 0,1,2,3,4,5,6,7,8,9,10,11,12,13,14,15);
}

// ---------------------------------------------------------------------------
// Kernel 1: brute-force 3-NN per dst point (src positions staged in LDS)
// grid = 128 blocks x 256 threads; 32 blocks per batch
// ---------------------------------------------------------------------------
__global__ void knn_kernel(const float* __restrict__ pos,
                           const float* __restrict__ pos_skip,
                           int* __restrict__ knn_idx,
                           float* __restrict__ knn_w) {
    __shared__ float sp[N_SRC_ * 3];
    const int b = blockIdx.x >> 5;                    // 32 blocks per batch
    const int i = blockIdx.x * 256 + threadIdx.x;     // global dst row

    const float* srcb = pos + (size_t)b * N_SRC_ * 3;
    for (int j = threadIdx.x; j < N_SRC_ * 3; j += 256) sp[j] = srcb[j];
    __syncthreads();

    const float px = pos_skip[(size_t)i * 3 + 0];
    const float py = pos_skip[(size_t)i * 3 + 1];
    const float pz = pos_skip[(size_t)i * 3 + 2];

    float d0 = 3.4e38f, d1 = 3.4e38f, d2 = 3.4e38f;
    int   i0 = 0, i1 = 0, i2 = 0;
    for (int j = 0; j < N_SRC_; ++j) {
        const float dx = sp[3 * j + 0] - px;
        const float dy = sp[3 * j + 1] - py;
        const float dz = sp[3 * j + 2] - pz;
        const float d  = dx * dx + dy * dy + dz * dz;
        if (d < d2) {
            if (d < d0)      { d2 = d1; i2 = i1; d1 = d0; i1 = i0; d0 = d; i0 = j; }
            else if (d < d1) { d2 = d1; i2 = i1; d1 = d;  i1 = j; }
            else             { d2 = d;  i2 = j; }
        }
    }
    const float w0 = 1.0f / fmaxf(d0, 1e-16f);
    const float w1 = 1.0f / fmaxf(d1, 1e-16f);
    const float w2 = 1.0f / fmaxf(d2, 1e-16f);
    const float inv = 1.0f / (w0 + w1 + w2);
    knn_idx[3 * i + 0] = i0;  knn_w[3 * i + 0] = w0 * inv;
    knn_idx[3 * i + 1] = i1;  knn_w[3 * i + 1] = w1 * inv;
    knn_idx[3 * i + 2] = i2;  knn_w[3 * i + 2] = w2 * inv;
}

// ---------------------------------------------------------------------------
// Kernel 2: build f16 activation matrix  A[i][0:256] = knn-interp(x),
//           A[i][256:384] = x_skip[i].   grid = 32768 blocks x 128 threads
// ---------------------------------------------------------------------------
__global__ void build_a_kernel(const float* __restrict__ x,
                               const float* __restrict__ x_skip,
                               const int* __restrict__ knn_idx,
                               const float* __restrict__ knn_w,
                               half_t* __restrict__ A) {
    const int i = blockIdx.x;
    const int t = threadIdx.x;           // 0..127
    const int b = i >> 13;               // i / 8192

    const int   j0 = knn_idx[3 * i + 0], j1 = knn_idx[3 * i + 1], j2 = knn_idx[3 * i + 2];
    const float w0 = knn_w[3 * i + 0],  w1 = knn_w[3 * i + 1],  w2 = knn_w[3 * i + 2];

    const float* r0 = x + (size_t)(b * N_SRC_ + j0) * C_IN_;
    const float* r1 = x + (size_t)(b * N_SRC_ + j1) * C_IN_;
    const float* r2 = x + (size_t)(b * N_SRC_ + j2) * C_IN_;
    half_t* arow = A + (size_t)i * K1_;

    arow[t]       = (half_t)(w0 * r0[t]       + w1 * r1[t]       + w2 * r2[t]);
    arow[t + 128] = (half_t)(w0 * r0[t + 128] + w1 * r1[t + 128] + w2 * r2[t + 128]);
    arow[t + 256] = (half_t)x_skip[(size_t)i * C_SKIP_ + t];
}

// ---------------------------------------------------------------------------
// Kernel 3: transpose + convert weights to f16  (W1t[N][K], W2t[N][K])
// ---------------------------------------------------------------------------
__global__ void conv_w_kernel(const float* __restrict__ W1, const float* __restrict__ W2,
                              half_t* __restrict__ W1t, half_t* __restrict__ W2t) {
    const int i = blockIdx.x * 256 + threadIdx.x;
    if (i < K1_ * H_) {                  // W1: [384][256] -> W1t: [256][384]
        const int k = i / H_, n = i % H_;
        W1t[(size_t)n * K1_ + k] = (half_t)W1[i];
    }
    if (i < H_ * C_OUT_) {               // W2: [256][128] -> W2t: [128][256]
        const int k = i / C_OUT_, n = i % C_OUT_;
        W2t[(size_t)n * H_ + k] = (half_t)W2[i];
    }
}

// ---------------------------------------------------------------------------
// Kernel 4: gate = relu(par @ Wp + bp), f32.  1 block x 512 threads
// ---------------------------------------------------------------------------
__global__ void gate_kernel(const float* __restrict__ par, const float* __restrict__ Wp,
                            const float* __restrict__ bp, float* __restrict__ gate) {
    const int t = threadIdx.x;           // 0..511
    const int b = t >> 7, j = t & 127;
    float s = bp[j];
#pragma unroll
    for (int p = 0; p < P_; ++p) s += par[b * P_ + p] * Wp[p * C_OUT_ + j];
    gate[t] = fmaxf(s, 0.0f);
}

// ---------------------------------------------------------------------------
// Kernel 5: fused MLP via WMMA, 2x4 / 2x2 register blocking for operand reuse.
//   GEMM1: [64 x 384] x [384 x 256] -> relu -> f16 LDS tile
//   GEMM2: [64 x 256] x [256 x 128] -> +b2, *gate -> f32 out
// grid = 512 blocks x 256 threads (8 waves)
// ---------------------------------------------------------------------------
__global__ void __launch_bounds__(256)
mlp_kernel(const half_t* __restrict__ A, const half_t* __restrict__ W1t,
           const half_t* __restrict__ W2t, const float* __restrict__ b1,
           const float* __restrict__ b2, const float* __restrict__ gate,
           float* __restrict__ out) {
    __shared__ half_t hbuf[64][LDH_];    // f16 hidden tile, 528B row stride

    const int tid  = threadIdx.x;
    const int wave = tid >> 5;
    const int lane = tid & 31;
    const int m    = lane & 15;          // row (A) / col (B) within 16-tile
    const int hi   = lane >> 4;
    const int koff = hi * 8;             // per-lane K offset within fragment

    const int row0 = blockIdx.x * 64;

    // Warm L2/L0 for the (shared, hot) weight matrices — global_prefetch_b8
    __builtin_prefetch(W1t + (size_t)tid * K1_, 0, 1);          // 256 rows
    __builtin_prefetch(W2t + (size_t)(tid & 127) * H_, 0, 1);   // 128 rows

    // ---------------- GEMM1: h = relu(A @ W1 + b1) ----------------
    // Wave (wm, wn): rows [wm*32, +32) (2 m-tiles), cols [wn*64, +64) (4 n-tiles).
    {
        const int wm = wave >> 2;        // 0..1
        const int wn = wave & 3;         // 0..3
        const half_t* arow0 = A + (size_t)(row0 + wm * 32 + m)      * K1_;
        const half_t* arow1 = A + (size_t)(row0 + wm * 32 + 16 + m) * K1_;

        v8f acc[2][4] = {};
        for (int k0 = 0; k0 < K1_; k0 += 32) {
            const v16h a0 = load_frag16(arow0 + k0 + koff);
            const v16h a1 = load_frag16(arow1 + k0 + koff);
            v16h bb[4];
#pragma unroll
            for (int ni = 0; ni < 4; ++ni)
                bb[ni] = load_frag16(W1t + (size_t)(wn * 64 + ni * 16 + m) * K1_ + k0 + koff);
#pragma unroll
            for (int ni = 0; ni < 4; ++ni) {
                acc[0][ni] = __builtin_amdgcn_wmma_f32_16x16x32_f16(
                    false, a0, false, bb[ni], (short)0, acc[0][ni], false, false);
                acc[1][ni] = __builtin_amdgcn_wmma_f32_16x16x32_f16(
                    false, a1, false, bb[ni], (short)0, acc[1][ni], false, false);
            }
        }
#pragma unroll
        for (int mi = 0; mi < 2; ++mi) {
#pragma unroll
            for (int ni = 0; ni < 4; ++ni) {
                const int colb = wn * 64 + ni * 16 + m;
                const float bias = b1[colb];
#pragma unroll
                for (int r = 0; r < 8; ++r) {
                    const int mloc = wm * 32 + mi * 16 + r + hi * 8;
                    const float v = acc[mi][ni][r] + bias;
                    hbuf[mloc][colb] = (half_t)(v > 0.0f ? v : 0.0f);
                }
            }
        }
    }
    __syncthreads();

    // ---------------- GEMM2: out = (h @ W2 + b2) * gate ----------------
    // Wave (wm2, wn2): rows [wm2*32, +32) (2 m-tiles), cols [wn2*32, +32) (2 n-tiles).
    {
        const int wm2 = wave >> 2;       // 0..1
        const int wn2 = wave & 3;        // 0..3
        v8f acc[2][2] = {};
        for (int k0 = 0; k0 < H_; k0 += 32) {
            const v16h a0 = load_frag16(&hbuf[wm2 * 32 + m][k0 + koff]);
            const v16h a1 = load_frag16(&hbuf[wm2 * 32 + 16 + m][k0 + koff]);
            v16h bb[2];
#pragma unroll
            for (int ni = 0; ni < 2; ++ni)
                bb[ni] = load_frag16(W2t + (size_t)(wn2 * 32 + ni * 16 + m) * H_ + k0 + koff);
#pragma unroll
            for (int ni = 0; ni < 2; ++ni) {
                acc[0][ni] = __builtin_amdgcn_wmma_f32_16x16x32_f16(
                    false, a0, false, bb[ni], (short)0, acc[0][ni], false, false);
                acc[1][ni] = __builtin_amdgcn_wmma_f32_16x16x32_f16(
                    false, a1, false, bb[ni], (short)0, acc[1][ni], false, false);
            }
        }
        const int batch = blockIdx.x >> 7;   // 128 blocks per batch
#pragma unroll
        for (int mi = 0; mi < 2; ++mi) {
#pragma unroll
            for (int ni = 0; ni < 2; ++ni) {
                const int col = wn2 * 32 + ni * 16 + m;
                const float g  = gate[batch * C_OUT_ + col];
                const float bb2 = b2[col];
#pragma unroll
                for (int r = 0; r < 8; ++r) {
                    const int grow = row0 + wm2 * 32 + mi * 16 + r + hi * 8;
                    out[(size_t)grow * C_OUT_ + col] = (acc[mi][ni][r] + bb2) * g;
                }
            }
        }
    }
}

// ---------------------------------------------------------------------------
// Host launcher
// ---------------------------------------------------------------------------
extern "C" void kernel_launch(void* const* d_in, const int* in_sizes, int n_in,
                              void* d_out, int out_size, void* d_ws, size_t ws_size,
                              hipStream_t stream) {
    (void)in_sizes; (void)n_in; (void)out_size; (void)ws_size;

    const float* par      = (const float*)d_in[0];
    const float* x        = (const float*)d_in[1];
    const float* pos      = (const float*)d_in[2];
    const float* x_skip   = (const float*)d_in[3];
    const float* pos_skip = (const float*)d_in[4];
    const float* W1       = (const float*)d_in[5];
    const float* b1       = (const float*)d_in[6];
    const float* W2       = (const float*)d_in[7];
    const float* b2       = (const float*)d_in[8];
    const float* Wp       = (const float*)d_in[9];
    const float* bp       = (const float*)d_in[10];
    // d_in[11] = batch (unused), d_in[12] = batch_skip (copied through)

    // Workspace layout (all offsets 256B-aligned)
    char* ws = (char*)d_ws;
    size_t off = 0;
    half_t* Amat  = (half_t*)(ws + off); off += (size_t)M_TOTAL * K1_ * sizeof(half_t);   // 24 MB
    half_t* W1t   = (half_t*)(ws + off); off += (size_t)H_ * K1_ * sizeof(half_t);        // 192 KB
    half_t* W2t   = (half_t*)(ws + off); off += (size_t)C_OUT_ * H_ * sizeof(half_t);     // 64 KB
    float*  gate  = (float*)(ws + off);  off += (size_t)B_ * C_OUT_ * sizeof(float);      // 2 KB
    int*    kidx  = (int*)(ws + off);    off += (size_t)M_TOTAL * 3 * sizeof(int);
    float*  kw    = (float*)(ws + off);  off += (size_t)M_TOTAL * 3 * sizeof(float);

    float* out = (float*)d_out;

    conv_w_kernel<<<(K1_ * H_ + 255) / 256, 256, 0, stream>>>(W1, W2, W1t, W2t);
    gate_kernel<<<1, 512, 0, stream>>>(par, Wp, bp, gate);
    knn_kernel<<<M_TOTAL / 256, 256, 0, stream>>>(pos, pos_skip, kidx, kw);
    build_a_kernel<<<M_TOTAL, 128, 0, stream>>>(x, x_skip, kidx, kw, Amat);
    mlp_kernel<<<M_TOTAL / 64, 256, 0, stream>>>(Amat, W1t, W2t, b1, b2, gate, out);

    // Tuple outputs 2 & 3: pass-through copies (pos_skip f32, batch_skip i64)
    hipMemcpyAsync(out + (size_t)M_TOTAL * C_OUT_, pos_skip,
                   (size_t)M_TOTAL * 3 * sizeof(float), hipMemcpyDeviceToDevice, stream);
    hipMemcpyAsync(out + (size_t)M_TOTAL * C_OUT_ + (size_t)M_TOTAL * 3, d_in[12],
                   (size_t)M_TOTAL * sizeof(long long), hipMemcpyDeviceToDevice, stream);
}
